// RNN_LSTM_truncated_embed_90907277787357
// MI455X (gfx1250) — matrix operand order repaired and checked by reading, not compiled, not run on hardware
//
#include <hip/hip_runtime.h>
#include <math.h>

typedef __attribute__((ext_vector_type(16))) __bf16 v16bf;
typedef __attribute__((ext_vector_type(8)))  __bf16 v8bf;
typedef __attribute__((ext_vector_type(8)))  float  v8f;

#define B_   64
#define T_   512
#define E_   256
#define H_   768
#define V_   5000
#define G4_  (4 * H_)     // 3072 gate columns
#define KC_  (E_ + H_)    // 1024 concat K for recurrent GEMM
#define VPAD 5056         // V padded to multiple of 64

// ---------------------------------------------------------------------------
// Fragment loaders: contiguous K order used by both A and B (not the ISA table order)
// A 16x32 (MxK): lane L: M = L%16, g = L/16; elems 0..7 -> K = k0+16g+i,
//                elems 8..15 -> K = k0+16g+8+i  (two 16B runs)
// B 32x16 (KxN): lane L: N = L%16, g = L/16; elems 0..15 -> K = k0+16g+i
//                (one contiguous 32B run; weights stored row-major [N][K])
// ---------------------------------------------------------------------------
__device__ __forceinline__ v16bf load_a_frag(const __bf16* __restrict__ base,
                                             int ldk, int mBase, int k0, int lane) {
  const __bf16* row = base + (size_t)(mBase + (lane & 15)) * ldk;
  int g = lane >> 4;
  v8bf lo = *(const v8bf*)(row + k0 + 16 * g);
  v8bf hi = *(const v8bf*)(row + k0 + 16 * g + 8);
  v16bf a;
#pragma unroll
  for (int i = 0; i < 8; ++i) { a[i] = lo[i]; a[8 + i] = hi[i]; }
  return a;
}

__device__ __forceinline__ v16bf load_b_frag(const __bf16* __restrict__ base,
                                             int ldk, int nBase, int k0, int lane) {
  const __bf16* row = base + (size_t)(nBase + (lane & 15)) * ldk;
  int g = lane >> 4;
  return *(const v16bf*)(row + k0 + 16 * g);
}

// ---------------------------------------------------------------------------
// Prep kernels
// ---------------------------------------------------------------------------
__global__ void k_embed(const int* __restrict__ x, const float* __restrict__ embed,
                        __bf16* __restrict__ xe) {
  size_t idx = (size_t)blockIdx.x * blockDim.x + threadIdx.x;   // 4 elems/thread
  size_t total = (size_t)T_ * B_ * (E_ / 4);
  if (idx >= total) return;
  int e4 = (int)(idx % (E_ / 4));
  size_t tb = idx / (E_ / 4);
  int b = (int)(tb % B_);
  int t = (int)(tb / B_);
  int tok = x[(size_t)b * T_ + t];
  float4 f = *(const float4*)(embed + (size_t)tok * E_ + e4 * 4);
  __bf16* dst = xe + ((size_t)t * B_ + b) * E_ + e4 * 4;
  dst[0] = (__bf16)f.x; dst[1] = (__bf16)f.y;
  dst[2] = (__bf16)f.z; dst[3] = (__bf16)f.w;
}

__global__ void k_prep_wcat(const float* __restrict__ Wih, const float* __restrict__ Whh,
                            __bf16* __restrict__ wcat) {
  size_t idx = (size_t)blockIdx.x * blockDim.x + threadIdx.x;   // 4 elems/thread
  size_t total = (size_t)G4_ * (KC_ / 4);
  if (idx >= total) return;
  int k = (int)(idx % (KC_ / 4)) * 4;
  size_t j = idx / (KC_ / 4);
  const float* src = (k < E_) ? (Wih + j * E_ + k) : (Whh + j * H_ + (k - E_));
  float4 f = *(const float4*)src;
  __bf16* dst = wcat + j * KC_ + k;
  dst[0] = (__bf16)f.x; dst[1] = (__bf16)f.y;
  dst[2] = (__bf16)f.z; dst[3] = (__bf16)f.w;
}

__global__ void k_prep_fcw(const float* __restrict__ fcw, __bf16* __restrict__ fcp) {
  size_t idx = (size_t)blockIdx.x * blockDim.x + threadIdx.x;   // 4 elems/thread
  size_t total = (size_t)VPAD * (H_ / 4);
  if (idx >= total) return;
  int h = (int)(idx % (H_ / 4)) * 4;
  size_t v = idx / (H_ / 4);
  __bf16* dst = fcp + v * H_ + h;
  if (v < V_) {
    float4 f = *(const float4*)(fcw + v * H_ + h);
    dst[0] = (__bf16)f.x; dst[1] = (__bf16)f.y;
    dst[2] = (__bf16)f.z; dst[3] = (__bf16)f.w;
  } else {
    dst[0] = (__bf16)0.f; dst[1] = (__bf16)0.f;
    dst[2] = (__bf16)0.f; dst[3] = (__bf16)0.f;
  }
}

__global__ void k_init(float* __restrict__ c, __bf16* __restrict__ h0,
                       __bf16* __restrict__ h1) {
  int idx = blockIdx.x * blockDim.x + threadIdx.x;
  if (idx >= B_ * H_) return;
  c[idx] = 0.f;
  h0[idx] = (__bf16)0.f;
  h1[idx] = (__bf16)0.f;
}

// ---------------------------------------------------------------------------
// One LSTM time step. grid = 96 blocks (48 j-blocks x 2 m-halves), 256
// threads (8 waves), ONE 16x16 gate tile per wave -> max parallelism per
// step, and two independent WMMA accumulator chains per tile (even/odd
// 32-wide K chunks) to cut the dependent-WMMA chain in half.
// Block covers 32 batch rows x 16 hidden cols across all 4 gates, so the
// fused fp32 elementwise LSTM update happens after one LDS exchange.
// ---------------------------------------------------------------------------
__global__ void __launch_bounds__(256)
k_lstm_step(int t,
            const __bf16* __restrict__ xe,    // [T][B][E]
            const __bf16* __restrict__ hin,   // [B][H]  bf16 (prev step)
            __bf16* __restrict__ hout,        // [B][H]  bf16 (this step)
            float* __restrict__ c,            // [B][H]  fp32 state
            __bf16* __restrict__ hs,          // [T][B][H] bf16 history
            const __bf16* __restrict__ wcat,  // [4H][KC] = [W_ih | W_hh]
            const float* __restrict__ b_ih,
            const float* __restrict__ b_hh) {
  __shared__ float gates[32][65];
  const int lane  = threadIdx.x & 31;
  const int wave  = threadIdx.x >> 5;
  const int nb    = blockIdx.x >> 1;            // hidden col block 0..47
  const int mhalf = blockIdx.x & 1;             // batch half 0..1
  const __bf16* xet = xe + (size_t)t * B_ * E_;

  const int gate = wave >> 1;                   // i/f/g/o
  const int mt2  = wave & 1;                    // m-tile within half
  const int mB   = (mhalf * 2 + mt2) * 16;      // global batch row base
  const int nGlb = gate * H_ + nb * 16;         // row in wcat

  v8f acc0 = {}, acc1 = {};
#pragma unroll
  for (int kk = 0; kk < E_; kk += 64) {         // x-part: 4 chunk-pairs
    v16bf a0 = load_a_frag(xet, E_, mB, kk,      lane);
    v16bf b0 = load_b_frag(wcat, KC_, nGlb, kk,      lane);
    v16bf a1 = load_a_frag(xet, E_, mB, kk + 32, lane);
    v16bf b1 = load_b_frag(wcat, KC_, nGlb, kk + 32, lane);
    acc0 = __builtin_amdgcn_wmma_f32_16x16x32_bf16(false, a0, false, b0,
                                                   (short)0, acc0, false, false);
    acc1 = __builtin_amdgcn_wmma_f32_16x16x32_bf16(false, a1, false, b1,
                                                   (short)0, acc1, false, false);
  }
#pragma unroll 4
  for (int kk = 0; kk < H_; kk += 64) {         // h-part: 12 chunk-pairs
    v16bf a0 = load_a_frag(hin, H_, mB, kk,      lane);
    v16bf b0 = load_b_frag(wcat, KC_, nGlb, E_ + kk,      lane);
    v16bf a1 = load_a_frag(hin, H_, mB, kk + 32, lane);
    v16bf b1 = load_b_frag(wcat, KC_, nGlb, E_ + kk + 32, lane);
    acc0 = __builtin_amdgcn_wmma_f32_16x16x32_bf16(false, a0, false, b0,
                                                   (short)0, acc0, false, false);
    acc1 = __builtin_amdgcn_wmma_f32_16x16x32_bf16(false, a1, false, b1,
                                                   (short)0, acc1, false, false);
  }

  // C/D layout: VGPR r, lane L: M = r + 8*(L/16), N = L%16
  int g = lane >> 4, nl = lane & 15;
#pragma unroll
  for (int r = 0; r < 8; ++r)
    gates[mt2 * 16 + g * 8 + r][gate * 16 + nl] = acc0[r] + acc1[r];
  __syncthreads();

  // elementwise LSTM: 32 rows x 16 cols = 512 elements, 2 per thread
#pragma unroll
  for (int it = 0; it < 2; ++it) {
    int el = threadIdx.x + it * 256;
    int ml = el >> 4;                           // local row 0..31
    int nl2 = el & 15;
    int m  = mhalf * 32 + ml;                   // batch row
    int j  = nb * 16 + nl2;                     // hidden index
    float ig = gates[ml][nl2]      + b_ih[j]          + b_hh[j];
    float fg = gates[ml][16 + nl2] + b_ih[H_ + j]     + b_hh[H_ + j];
    float gg = gates[ml][32 + nl2] + b_ih[2 * H_ + j] + b_hh[2 * H_ + j];
    float og = gates[ml][48 + nl2] + b_ih[3 * H_ + j] + b_hh[3 * H_ + j];
    float si = 1.f / (1.f + __expf(-ig));
    float sf = 1.f / (1.f + __expf(-fg));
    float so = 1.f / (1.f + __expf(-og));
    float tg = tanhf(gg);
    size_t off = (size_t)m * H_ + j;
    float cn = sf * c[off] + si * tg;
    float hn = so * tanhf(cn);
    c[off] = cn;
    __bf16 hb = (__bf16)hn;
    hout[off] = hb;
    hs[(size_t)t * B_ * H_ + off] = hb;
  }
}

// ---------------------------------------------------------------------------
// Output projection: [T*B, H] @ [H, V] + fc_b. Each wave: 2 m-tiles x 4
// n-tiles (A and B fragment reuse: 6 fragment loads feed 8 WMMAs), K = 768.
// 8 independent accumulator chains for ILP.
// ---------------------------------------------------------------------------
__global__ void __launch_bounds__(256)
k_out_gemm(const __bf16* __restrict__ hs,     // [T*B][H] row-major
           const __bf16* __restrict__ fcp,    // [VPAD][H] row-major (bf16)
           const float* __restrict__ fc_b,
           float* __restrict__ out) {         // [B][T][V] fp32
  const int lane = threadIdx.x & 31;
  const int wave = threadIdx.x >> 5;
  const int NG   = VPAD / 64;                 // 79 n-groups
  size_t job = (size_t)blockIdx.x * 8 + wave; // 1024 m-pairs * 79 n-groups
  int ng = (int)(job % NG);
  int mp = (int)(job / NG);
  int mBase = mp * 32;
  int nBase = ng * 64;

  v8f acc[2][4] = {};
#pragma unroll 2
  for (int kk = 0; kk < H_; kk += 32) {
    v16bf a0 = load_a_frag(hs, H_, mBase,      kk, lane);
    v16bf a1 = load_a_frag(hs, H_, mBase + 16, kk, lane);
#pragma unroll
    for (int n = 0; n < 4; ++n) {
      v16bf b = load_b_frag(fcp, H_, nBase + n * 16, kk, lane);
      acc[0][n] = __builtin_amdgcn_wmma_f32_16x16x32_bf16(false, a0, false, b,
                                                          (short)0, acc[0][n],
                                                          false, false);
      acc[1][n] = __builtin_amdgcn_wmma_f32_16x16x32_bf16(false, a1, false, b,
                                                          (short)0, acc[1][n],
                                                          false, false);
    }
  }

  int g = lane >> 4, nl = lane & 15;
#pragma unroll
  for (int n = 0; n < 4; ++n) {
    int v = nBase + n * 16 + nl;
    if (v < V_) {
      float bias = fc_b[v];
#pragma unroll
      for (int mi = 0; mi < 2; ++mi) {
#pragma unroll
        for (int r = 0; r < 8; ++r) {
          int m = mBase + mi * 16 + g * 8 + r;  // row = t*64 + b
          int tt = m >> 6, bb = m & 63;
          out[(size_t)bb * T_ * V_ + (size_t)tt * V_ + v] = acc[mi][n][r] + bias;
        }
      }
    }
  }
}

// ---------------------------------------------------------------------------
// Host side
// ---------------------------------------------------------------------------
extern "C" void kernel_launch(void* const* d_in, const int* in_sizes, int n_in,
                              void* d_out, int out_size, void* d_ws, size_t ws_size,
                              hipStream_t stream) {
  const int*   x     = (const int*)d_in[0];
  // d_in[1] = truncate_length (forward-invariant, unused)
  const float* embed = (const float*)d_in[2];
  const float* W_ih  = (const float*)d_in[3];
  const float* W_hh  = (const float*)d_in[4];
  const float* b_ih  = (const float*)d_in[5];
  const float* b_hh  = (const float*)d_in[6];
  const float* fc_w  = (const float*)d_in[7];
  const float* fc_b  = (const float*)d_in[8];
  float* out = (float*)d_out;

  char* ws = (char*)d_ws;
  size_t o = 0;
  auto alloc = [&](size_t n) -> char* {
    char* p = ws + o;
    o = (o + n + 255) & ~(size_t)255;
    return p;
  };
  __bf16* xe    = (__bf16*)alloc((size_t)T_ * B_ * E_ * 2);   // 16 MB
  __bf16* hs    = (__bf16*)alloc((size_t)T_ * B_ * H_ * 2);   // 48 MB
  __bf16* hbuf0 = (__bf16*)alloc((size_t)B_ * H_ * 2);
  __bf16* hbuf1 = (__bf16*)alloc((size_t)B_ * H_ * 2);
  float*  cbuf  = (float*) alloc((size_t)B_ * H_ * 4);
  __bf16* wcat  = (__bf16*)alloc((size_t)G4_ * KC_ * 2);      // 6 MB
  __bf16* fcp   = (__bf16*)alloc((size_t)VPAD * H_ * 2);      // 7.7 MB

  k_embed<<<(T_ * B_ * (E_ / 4) + 255) / 256, 256, 0, stream>>>(x, embed, xe);
  k_prep_wcat<<<(G4_ * (KC_ / 4) + 255) / 256, 256, 0, stream>>>(W_ih, W_hh, wcat);
  k_prep_fcw<<<(VPAD * (H_ / 4) + 255) / 256, 256, 0, stream>>>(fc_w, fcp);
  k_init<<<(B_ * H_ + 255) / 256, 256, 0, stream>>>(cbuf, hbuf0, hbuf1);

  for (int t = 0; t < T_; ++t) {
    const __bf16* hin = (t & 1) ? hbuf1 : hbuf0;
    __bf16*       hho = (t & 1) ? hbuf0 : hbuf1;
    k_lstm_step<<<96, 256, 0, stream>>>(t, xe, hin, hho, cbuf, hs, wcat,
                                        b_ih, b_hh);
  }

  int jobs = (T_ * B_ / 32) * (VPAD / 64);   // 1024 * 79 = 80896
  k_out_gemm<<<jobs / 8, 256, 0, stream>>>(hs, fcp, fc_b, out);
}
